// HeteroActorCriticEncoder_31267361915426
// MI455X (gfx1250) — compile-verified, matrix operand order checked by reading, not run on hardware
//
#include <hip/hip_runtime.h>
#include <math.h>

// ---------------- problem constants ----------------
#define NP 100000          // N_PLACE
#define NT 100000          // N_TRANS
#define EE 1000000         // edges per relation
#define FI 128
#define FO 64
#define NH 4
#define HD 16
#define NEG_SLOPE 0.2f
#define EPSV 1e-16f

typedef __attribute__((ext_vector_type(16))) _Float16 v16h;
typedef __attribute__((ext_vector_type(8)))  _Float16 v8h;
typedef __attribute__((ext_vector_type(8)))  float    v8f;

// ---------------- workspace layout (units: 4-byte words) ----------------
// hp/ht are stored as f16 (half the region is used; offsets kept word-based)
#define OFF_HP       ((size_t)0)
#define OFF_HT       (OFF_HP + (size_t)NP * 64)
#define OFF_ALS_PT   (OFF_HT + (size_t)NT * 64)        // place as src in p->t
#define OFF_ALS_PP   (OFF_ALS_PT + (size_t)NP * 4)
#define OFF_ALD_TP   (OFF_ALS_PP + (size_t)NP * 4)     // place as dst in t->p
#define OFF_ALD_PP   (OFF_ALD_TP + (size_t)NP * 4)
#define OFF_ALS_TP   (OFF_ALD_PP + (size_t)NP * 4)     // trans as src in t->p
#define OFF_ALS_TT   (OFF_ALS_TP + (size_t)NT * 4)
#define OFF_ALD_PT   (OFF_ALS_TT + (size_t)NT * 4)     // trans as dst in p->t
#define OFF_ALD_TT   (OFF_ALD_PT + (size_t)NT * 4)
#define OFF_ZERO     (OFF_ALD_TT + (size_t)NT * 4)     // ---- zeroed every call ----
#define OFF_M_PT     (OFF_ZERO)                        // uint max-keys
#define OFF_M_TT     (OFF_M_PT + (size_t)NT * 4)
#define OFF_M_TP     (OFF_M_TT + (size_t)NT * 4)
#define OFF_M_PP     (OFF_M_TP + (size_t)NP * 4)
#define OFF_S_PT     (OFF_M_PP + (size_t)NP * 4)
#define OFF_S_TT     (OFF_S_PT + (size_t)NT * 4)
#define OFF_S_TP     (OFF_S_TT + (size_t)NT * 4)
#define OFF_S_PP     (OFF_S_TP + (size_t)NP * 4)
#define OFF_ACC_T_PT (OFF_S_PP + (size_t)NP * 4)
#define OFF_ACC_T_TT (OFF_ACC_T_PT + (size_t)NT * 64)
#define OFF_ACC_P_TP (OFF_ACC_T_TT + (size_t)NT * 64)
#define OFF_ACC_P_PP (OFF_ACC_P_TP + (size_t)NP * 64)
#define OFF_SCORE    (OFF_ACC_P_PP + (size_t)NP * 64)  // 4 relations x 64
#define OFF_ATTN     (OFF_SCORE + (size_t)256)
#define OFF_END      (OFF_ATTN + (size_t)4)

#define SW_PITCH (FI + 8)   // padded LDS pitch (halves) for W  [n][k]
#define SK_PITCH (FO + 8)   // padded LDS pitch (halves) for kW [n][k]

// monotonic float <-> uint key (for atomicMax on signed floats; key 0 == -inf)
__device__ __forceinline__ unsigned f2key(float f) {
  int i = __float_as_int(f);
  return (i >= 0) ? ((unsigned)i | 0x80000000u) : ~(unsigned)i;
}
__device__ __forceinline__ float key2f(unsigned k) {
  int i = (k & 0x80000000u) ? (int)(k & 0x7FFFFFFFu) : (int)~k;
  return __int_as_float(i);
}

// ---------------- WMMA projection GEMM: H16 = f16(X[N,128] * W[128,64] + b) -------
// W staged once per workgroup into LDS (f16, transposed [n][k], padded) so each
// lane's B fragment is two ds_load_b128. A fragments via global_load_b128.
__global__ void proj_wmma_kernel(const float* __restrict__ X,
                                 const float* __restrict__ W,
                                 const float* __restrict__ bias,
                                 _Float16* __restrict__ H16,
                                 int nTiles) {
  __shared__ _Float16 sW[FO * SW_PITCH];
  for (int idx = threadIdx.x; idx < FI * FO; idx += blockDim.x) {
    int k = idx >> 6, n = idx & 63;
    sW[n * SW_PITCH + k] = (_Float16)W[idx];
  }
  __syncthreads();

  const int wave = threadIdx.x >> 5;
  const int tile = blockIdx.x * 4 + wave;
  if (tile >= nTiles) return;                 // wave-uniform: EXEC stays all-ones
  const int lane  = threadIdx.x & 31;
  const int half  = lane >> 4;
  const int r     = lane & 15;
  const int row   = tile * 16 + r;
  const int kbase = half * 8;

  v8f acc[4];
#pragma unroll
  for (int j = 0; j < 4; ++j)
#pragma unroll
    for (int i = 0; i < 8; ++i) acc[j][i] = 0.0f;

  for (int k0 = 0; k0 < FI; k0 += 32) {
    const float* xp = X + (size_t)row * FI + k0 + kbase;
    if (k0 + 32 < FI) __builtin_prefetch(xp + 32, 0, 0);   // global_prefetch_b8
    const float4* xp4 = (const float4*)xp;                  // 16B aligned (kbase mult of 8)
    float4 x0 = xp4[0], x1 = xp4[1];   // K = k0+kbase   .. +7
    float4 x2 = xp4[4], x3 = xp4[5];   // K = k0+16+kbase.. +7
    v16h a;
    a[0]=(_Float16)x0.x; a[1]=(_Float16)x0.y; a[2]=(_Float16)x0.z; a[3]=(_Float16)x0.w;
    a[4]=(_Float16)x1.x; a[5]=(_Float16)x1.y; a[6]=(_Float16)x1.z; a[7]=(_Float16)x1.w;
    a[8]=(_Float16)x2.x; a[9]=(_Float16)x2.y; a[10]=(_Float16)x2.z; a[11]=(_Float16)x2.w;
    a[12]=(_Float16)x3.x; a[13]=(_Float16)x3.y; a[14]=(_Float16)x3.z; a[15]=(_Float16)x3.w;
#pragma unroll
    for (int j = 0; j < 4; ++j) {
      const int n = j * 16 + r;
      const _Float16* wp = sW + n * SW_PITCH + k0 + kbase;  // 16B aligned
      v8h blo = *(const v8h*)wp;
      v8h bhi = *(const v8h*)(wp + 16);
      v16h b;
#pragma unroll
      for (int i = 0; i < 8; ++i) { b[i] = blo[i]; b[8 + i] = bhi[i]; }
      acc[j] = __builtin_amdgcn_wmma_f32_16x16x32_f16(
          false, a, false, b, (short)0, acc[j], false, false);
    }
  }
#pragma unroll
  for (int j = 0; j < 4; ++j) {
    const int n  = j * 16 + r;
    const float bv = bias[n];
#pragma unroll
    for (int m = 0; m < 8; ++m) {
      const int rowm = tile * 16 + m + half * 8;
      H16[(size_t)rowm * FO + n] = (_Float16)(acc[j][m] + bv);
    }
  }
}

// ---------------- per-node attention logits (4 dot products per (n,h)) ----------------
__global__ void logits_kernel(const _Float16* __restrict__ Hm, int N,
                              const float* __restrict__ aS0, const float* __restrict__ aS1,
                              const float* __restrict__ aD0, const float* __restrict__ aD1,
                              float* __restrict__ oS0, float* __restrict__ oS1,
                              float* __restrict__ oD0, float* __restrict__ oD1) {
  int t = blockIdx.x * blockDim.x + threadIdx.x;
  if (t >= N * NH) return;
  int n = t >> 2, h = t & 3;
  const _Float16* hp = Hm + (size_t)n * FO + h * HD;
  float s0 = 0.f, s1 = 0.f, d0 = 0.f, d1 = 0.f;
#pragma unroll
  for (int d = 0; d < HD; ++d) {
    float v = (float)hp[d];
    s0 += v * aS0[h * HD + d];
    s1 += v * aS1[h * HD + d];
    d0 += v * aD0[h * HD + d];
    d1 += v * aD1[h * HD + d];
  }
  oS0[t] = s0; oS1[t] = s1; oD0[t] = d0; oD1[t] = d1;
}

// ---------------- edge pass 1: segment max via monotonic-key atomicMax ----------------
__global__ void edge_max_kernel(const int* __restrict__ ei,
                                const float* __restrict__ alS,
                                const float* __restrict__ alD,
                                unsigned* __restrict__ mkey) {
  int t = blockIdx.x * blockDim.x + threadIdx.x;
  if (t >= EE * NH) return;
  int e = t >> 2, h = t & 3;
  int src = ei[e], dst = ei[EE + e];
  float a = alS[src * NH + h] + alD[dst * NH + h];
  a = (a > 0.f) ? a : a * NEG_SLOPE;
  atomicMax(&mkey[dst * NH + h], f2key(a));
}

// ---------------- edge pass 2: e = exp(alpha-m); scatter e and e*x_src ----------------
// Gathers the f16 copy of the projected features: halves the dominant random-read
// traffic (128 B/edge instead of 256 B/edge). Accumulation stays f32 via atomics.
__global__ void edge_acc_kernel(const int* __restrict__ ei,
                                const float* __restrict__ alS,
                                const float* __restrict__ alD,
                                const unsigned* __restrict__ mkey,
                                const _Float16* __restrict__ Hs,
                                float* __restrict__ sacc,
                                float* __restrict__ outacc) {
  int t = blockIdx.x * blockDim.x + threadIdx.x;
  if (t >= EE * FO) return;
  int e = t >> 6, f = t & 63, h = f >> 4;
  int src = ei[e], dst = ei[EE + e];
  float a = alS[src * NH + h] + alD[dst * NH + h];
  a = (a > 0.f) ? a : a * NEG_SLOPE;
  float m  = key2f(mkey[dst * NH + h]);
  float ex = __expf(a - m);
  if ((f & 15) == 0) atomicAdd(&sacc[dst * NH + h], ex);
  atomicAdd(&outacc[(size_t)dst * FO + f], ex * (float)Hs[(size_t)src * FO + f]);
}

// ---------------- per-node finalize: out = relu(acc / (s + eps)) ----------------
__global__ void finalize_kernel(float* __restrict__ outacc,
                                const float* __restrict__ sacc, int N) {
  int t = blockIdx.x * blockDim.x + threadIdx.x;
  if (t >= N * FO) return;
  int n = t >> 6, h = (t >> 4) & 3;
  float v = outacc[t] / (sacc[n * NH + h] + EPSV);
  outacc[t] = v > 0.f ? v : 0.f;
}

// ---------------- semantic score GEMM: colsum(tanh(M[N,64]*kW[64,64]+kb)) ----------------
__global__ void score_wmma_kernel(const float* __restrict__ M,
                                  const float* __restrict__ kW,
                                  const float* __restrict__ kb,
                                  float* __restrict__ score_acc,  // 64 floats (one relation)
                                  int nTiles) {
  __shared__ _Float16 sK[FO * SK_PITCH];
  for (int idx = threadIdx.x; idx < FO * FO; idx += blockDim.x) {
    int k = idx >> 6, n = idx & 63;
    sK[n * SK_PITCH + k] = (_Float16)kW[idx];
  }
  __syncthreads();

  const int wave = threadIdx.x >> 5;
  const int tile = blockIdx.x * 4 + wave;
  if (tile >= nTiles) return;
  const int lane  = threadIdx.x & 31;
  const int half  = lane >> 4;
  const int r     = lane & 15;
  const int row   = tile * 16 + r;
  const int kbase = half * 8;

  v8f acc[4];
#pragma unroll
  for (int j = 0; j < 4; ++j)
#pragma unroll
    for (int i = 0; i < 8; ++i) acc[j][i] = 0.0f;

#pragma unroll
  for (int k0 = 0; k0 < FO; k0 += 32) {
    const float4* xp4 = (const float4*)(M + (size_t)row * FO + k0 + kbase);
    float4 x0 = xp4[0], x1 = xp4[1];
    float4 x2 = xp4[4], x3 = xp4[5];
    v16h a;
    a[0]=(_Float16)x0.x; a[1]=(_Float16)x0.y; a[2]=(_Float16)x0.z; a[3]=(_Float16)x0.w;
    a[4]=(_Float16)x1.x; a[5]=(_Float16)x1.y; a[6]=(_Float16)x1.z; a[7]=(_Float16)x1.w;
    a[8]=(_Float16)x2.x; a[9]=(_Float16)x2.y; a[10]=(_Float16)x2.z; a[11]=(_Float16)x2.w;
    a[12]=(_Float16)x3.x; a[13]=(_Float16)x3.y; a[14]=(_Float16)x3.z; a[15]=(_Float16)x3.w;
#pragma unroll
    for (int j = 0; j < 4; ++j) {
      const int n = j * 16 + r;
      const _Float16* wp = sK + n * SK_PITCH + k0 + kbase;
      v8h blo = *(const v8h*)wp;
      v8h bhi = *(const v8h*)(wp + 16);
      v16h b;
#pragma unroll
      for (int i = 0; i < 8; ++i) { b[i] = blo[i]; b[8 + i] = bhi[i]; }
      acc[j] = __builtin_amdgcn_wmma_f32_16x16x32_f16(
          false, a, false, b, (short)0, acc[j], false, false);
    }
  }
#pragma unroll
  for (int j = 0; j < 4; ++j) {
    const int n = j * 16 + r;
    const float bv = kb[n];
    float cs = 0.f;
#pragma unroll
    for (int m = 0; m < 8; ++m) cs += tanhf(acc[j][m] + bv);
    atomicAdd(&score_acc[n], cs);
  }
}

// ---------------- semantic softmax over relations ----------------
__global__ void semantic_kernel(const float* __restrict__ score_acc,
                                const float* __restrict__ q,
                                float* __restrict__ attn) {
  if (blockIdx.x != 0 || threadIdx.x != 0) return;
  float s[4];
  for (int rr = 0; rr < 4; ++rr) {
    float a = 0.f;
    for (int f = 0; f < FO; ++f) a += q[f] * score_acc[rr * FO + f];
    s[rr] = a / (float)((rr < 2) ? NP : NT);   // mean over nodes
  }
  float mx = fmaxf(s[0], s[1]);
  float e0 = __expf(s[0] - mx), e1 = __expf(s[1] - mx);
  attn[0] = e0 / (e0 + e1); attn[1] = e1 / (e0 + e1);
  mx = fmaxf(s[2], s[3]);
  e0 = __expf(s[2] - mx); e1 = __expf(s[3] - mx);
  attn[2] = e0 / (e0 + e1); attn[3] = e1 / (e0 + e1);
}

// ---------------- weighted combine of the two relation outputs ----------------
__global__ void combine_kernel(const float* __restrict__ attn,
                               const float* __restrict__ A0,
                               const float* __restrict__ A1,
                               float* __restrict__ outp,
                               int n, int i0, int i1) {
  int t = blockIdx.x * blockDim.x + threadIdx.x;
  if (t >= n) return;
  outp[t] = attn[i0] * A0[t] + attn[i1] * A1[t];
}

extern "C" void kernel_launch(void* const* d_in, const int* in_sizes, int n_in,
                              void* d_out, int out_size, void* d_ws, size_t ws_size,
                              hipStream_t stream) {
  const float* x_place = (const float*)d_in[0];
  const float* x_trans = (const float*)d_in[1];
  const float* W_place = (const float*)d_in[2];
  const float* b_place = (const float*)d_in[3];
  const float* W_trans = (const float*)d_in[4];
  const float* b_trans = (const float*)d_in[5];
  const float* as_pt = (const float*)d_in[6];
  const float* ad_pt = (const float*)d_in[7];
  const float* as_tp = (const float*)d_in[8];
  const float* ad_tp = (const float*)d_in[9];
  const float* as_pp = (const float*)d_in[10];
  const float* ad_pp = (const float*)d_in[11];
  const float* as_tt = (const float*)d_in[12];
  const float* ad_tt = (const float*)d_in[13];
  const float* q     = (const float*)d_in[14];
  const float* kW    = (const float*)d_in[15];
  const float* kb    = (const float*)d_in[16];
  const int* ei_pt = (const int*)d_in[17];
  const int* ei_tp = (const int*)d_in[18];
  const int* ei_pp = (const int*)d_in[19];
  const int* ei_tt = (const int*)d_in[20];

  float* ws = (float*)d_ws;
  _Float16* hp = (_Float16*)(ws + OFF_HP);
  _Float16* ht = (_Float16*)(ws + OFF_HT);
  float* alS_pt = ws + OFF_ALS_PT;  float* alS_pp = ws + OFF_ALS_PP;
  float* alD_tp = ws + OFF_ALD_TP;  float* alD_pp = ws + OFF_ALD_PP;
  float* alS_tp = ws + OFF_ALS_TP;  float* alS_tt = ws + OFF_ALS_TT;
  float* alD_pt = ws + OFF_ALD_PT;  float* alD_tt = ws + OFF_ALD_TT;
  unsigned* m_pt = (unsigned*)(ws + OFF_M_PT);
  unsigned* m_tt = (unsigned*)(ws + OFF_M_TT);
  unsigned* m_tp = (unsigned*)(ws + OFF_M_TP);
  unsigned* m_pp = (unsigned*)(ws + OFF_M_PP);
  float* s_pt = ws + OFF_S_PT;  float* s_tt = ws + OFF_S_TT;
  float* s_tp = ws + OFF_S_TP;  float* s_pp = ws + OFF_S_PP;
  float* acc_t_pt = ws + OFF_ACC_T_PT;
  float* acc_t_tt = ws + OFF_ACC_T_TT;
  float* acc_p_tp = ws + OFF_ACC_P_TP;
  float* acc_p_pp = ws + OFF_ACC_P_PP;
  float* score = ws + OFF_SCORE;
  float* attn  = ws + OFF_ATTN;

  // zero the atomic-accumulation region (max keys, sums, outputs, scores, attn)
  hipMemsetAsync(ws + OFF_ZERO, 0, (OFF_END - OFF_ZERO) * sizeof(float), stream);

  // 1) projections (WMMA, LDS-staged B): hp = f16(x_place*W_place + b) ; ht likewise
  const int projTiles = NP / 16;                       // 6250, exact
  const int projGrid  = (projTiles + 3) / 4;
  proj_wmma_kernel<<<projGrid, 128, 0, stream>>>(x_place, W_place, b_place, hp, projTiles);
  proj_wmma_kernel<<<projGrid, 128, 0, stream>>>(x_trans, W_trans, b_trans, ht, NT / 16);

  // 2) per-node logits
  {
    int thr = NP * NH, g = (thr + 255) / 256;
    logits_kernel<<<g, 256, 0, stream>>>(hp, NP, as_pt, as_pp, ad_tp, ad_pp,
                                         alS_pt, alS_pp, alD_tp, alD_pp);
    thr = NT * NH; g = (thr + 255) / 256;
    logits_kernel<<<g, 256, 0, stream>>>(ht, NT, as_tp, as_tt, ad_pt, ad_tt,
                                         alS_tp, alS_tt, alD_pt, alD_tt);
  }

  // 3) four relations: max -> exp/scatter -> finalize
  const int gMax = (EE * NH + 255) / 256;
  const int gAcc = (EE * FO + 255) / 256;
  // p->t (dst = trans)
  edge_max_kernel<<<gMax, 256, 0, stream>>>(ei_pt, alS_pt, alD_pt, m_pt);
  edge_acc_kernel<<<gAcc, 256, 0, stream>>>(ei_pt, alS_pt, alD_pt, m_pt, hp, s_pt, acc_t_pt);
  finalize_kernel<<<(NT * FO + 255) / 256, 256, 0, stream>>>(acc_t_pt, s_pt, NT);
  // t->t
  edge_max_kernel<<<gMax, 256, 0, stream>>>(ei_tt, alS_tt, alD_tt, m_tt);
  edge_acc_kernel<<<gAcc, 256, 0, stream>>>(ei_tt, alS_tt, alD_tt, m_tt, ht, s_tt, acc_t_tt);
  finalize_kernel<<<(NT * FO + 255) / 256, 256, 0, stream>>>(acc_t_tt, s_tt, NT);
  // t->p (dst = place)
  edge_max_kernel<<<gMax, 256, 0, stream>>>(ei_tp, alS_tp, alD_tp, m_tp);
  edge_acc_kernel<<<gAcc, 256, 0, stream>>>(ei_tp, alS_tp, alD_tp, m_tp, ht, s_tp, acc_p_tp);
  finalize_kernel<<<(NP * FO + 255) / 256, 256, 0, stream>>>(acc_p_tp, s_tp, NP);
  // p->p
  edge_max_kernel<<<gMax, 256, 0, stream>>>(ei_pp, alS_pp, alD_pp, m_pp);
  edge_acc_kernel<<<gAcc, 256, 0, stream>>>(ei_pp, alS_pp, alD_pp, m_pp, hp, s_pp, acc_p_pp);
  finalize_kernel<<<(NP * FO + 255) / 256, 256, 0, stream>>>(acc_p_pp, s_pp, NP);

  // 4) semantic attention scores (WMMA GEMM + tanh + column sums)
  // score rows: 0 = p_tp, 1 = p_pp (place group); 2 = t_pt, 3 = t_tt (trans group)
  score_wmma_kernel<<<projGrid, 128, 0, stream>>>(acc_p_tp, kW, kb, score + 0 * FO, NP / 16);
  score_wmma_kernel<<<projGrid, 128, 0, stream>>>(acc_p_pp, kW, kb, score + 1 * FO, NP / 16);
  score_wmma_kernel<<<projGrid, 128, 0, stream>>>(acc_t_pt, kW, kb, score + 2 * FO, NT / 16);
  score_wmma_kernel<<<projGrid, 128, 0, stream>>>(acc_t_tt, kW, kb, score + 3 * FO, NT / 16);
  semantic_kernel<<<1, 32, 0, stream>>>(score, q, attn);

  // 5) combine into d_out: out_place first, then out_trans
  float* out_place = (float*)d_out;
  float* out_trans = out_place + (size_t)NP * FO;
  combine_kernel<<<(NP * FO + 255) / 256, 256, 0, stream>>>(attn, acc_p_tp, acc_p_pp,
                                                            out_place, NP * FO, 0, 1);
  combine_kernel<<<(NT * FO + 255) / 256, 256, 0, stream>>>(attn, acc_t_pt, acc_t_tt,
                                                            out_trans, NT * FO, 2, 3);
}
